// ExpertChoiceRouter_38482906972900
// MI455X (gfx1250) — matrix operand order repaired and checked by reading, not compiled, not run on hardware
//
#include <hip/hip_runtime.h>

#define HIDDEN 2048
#define NUM_EXPERTS 64
#define TOKENS 32768
#define CAP 1024

typedef __attribute__((ext_vector_type(16))) _Float16 v16h;
typedef __attribute__((ext_vector_type(8)))  float    v8f;

// ---------------------------------------------------------------------------
// Kernel 1: logits = hidden @ gate^T via v_wmma_f32_16x16x32_f16, then clip +
// softmax over 64 experts. Each wave computes 16 tokens x 64 experts
// (4 accumulator tiles). Block = 8 waves = 128 tokens.
// ---------------------------------------------------------------------------
__global__ __launch_bounds__(256) void gemm_softmax_kernel(
    const float* __restrict__ hidden, const float* __restrict__ gate,
    float* __restrict__ probs_te, float* __restrict__ probs_et)
{
  __shared__ float lds[8][16][NUM_EXPERTS];  // 32 KB

  const int tid    = threadIdx.x;
  const int wave   = tid >> 5;
  const int lane   = tid & 31;
  const int lane16 = lane & 15;
  const int hi     = lane >> 4;                 // lane half (0/1)
  const int rowBase = blockIdx.x * 128 + wave * 16;

  v8f acc[4] = {};

  const int   aRow   = rowBase + lane16;
  const float* aPtr  = hidden + (size_t)aRow * HIDDEN;
  const int   kbaseA = hi ? 16 : 0;             // A frag: K = base+{0..15}
  const int   kbaseB = hi ? 16 : 0;             // B frag: K = base+{0..15}

  for (int k = 0; k < HIDDEN; k += 32) {
    // ---- A fragment (16x32 f16): lane holds row M=lane16 ----
    v16h a;
#pragma unroll
    for (int j = 0; j < 4; ++j) {
      float2 lo = *(const float2*)(aPtr + k + kbaseA + 2 * j);
      float2 h2 = *(const float2*)(aPtr + k + kbaseA + 8 + 2 * j);
      a[2 * j]     = (_Float16)lo.x;
      a[2 * j + 1] = (_Float16)lo.y;
      a[8 + 2 * j]     = (_Float16)h2.x;
      a[8 + 2 * j + 1] = (_Float16)h2.y;
    }
    // ---- 4 expert tiles of 16 columns each ----
#pragma unroll
    for (int n = 0; n < 4; ++n) {
      const float* bPtr = gate + (size_t)(n * 16 + lane16) * HIDDEN + k + kbaseB;
      v16h b;
#pragma unroll
      for (int c = 0; c < 4; ++c) {
        float4 v = *(const float4*)(bPtr + 4 * c);
        b[4 * c]     = (_Float16)v.x;
        b[4 * c + 1] = (_Float16)v.y;
        b[4 * c + 2] = (_Float16)v.z;
        b[4 * c + 3] = (_Float16)v.w;
      }
      acc[n] = __builtin_amdgcn_wmma_f32_16x16x32_f16(
          /*neg_a=*/false, a, /*neg_b=*/false, b,
          /*c_mod=*/(short)0, acc[n], /*reuse_a=*/false, /*reuse_b=*/false);
    }
  }

  // C/D layout: VGPR r, lane -> M = r + 8*hi, N = lane16 (+16*tile)
#pragma unroll
  for (int n = 0; n < 4; ++n)
#pragma unroll
    for (int r = 0; r < 8; ++r)
      lds[wave][r + 8 * hi][n * 16 + lane16] = acc[n][r];
  __syncthreads();

  // Per-token clip + softmax: lanes 0..15 each own one token of this wave.
  if (lane < 16) {
    const int m = lane;
    const int t = rowBase + m;
    float vals[NUM_EXPERTS];
    float mx = -1e30f;
#pragma unroll
    for (int e = 0; e < NUM_EXPERTS; ++e) {
      float x = lds[wave][m][e];
      x = fminf(10.0f, fmaxf(-10.0f, x));
      vals[e] = x;
      mx = fmaxf(mx, x);
    }
    float sum = 0.0f;
#pragma unroll
    for (int e = 0; e < NUM_EXPERTS; ++e) {
      float p = __expf(vals[e] - mx);
      vals[e] = p;
      sum += p;
    }
    const float inv = 1.0f / sum;
#pragma unroll
    for (int e = 0; e < NUM_EXPERTS; ++e) {
      float p = vals[e] * inv;
      probs_te[(size_t)t * NUM_EXPERTS + e] = p;
      probs_et[(size_t)e * TOKENS + t]      = p;   // 16 consecutive t per store
    }
  }
}

// ---------------------------------------------------------------------------
// Kernel 2: per-expert exact 1024th-largest via 4-round byte radix select on
// float bit patterns (probs > 0 so uint order == float order).
// Outputs threshold theta[e] and tie quota[e].
// ---------------------------------------------------------------------------
__global__ __launch_bounds__(256) void threshold_kernel(
    const float* __restrict__ probs_et,
    unsigned* __restrict__ thr, unsigned* __restrict__ quota)
{
  __shared__ unsigned hist[256];
  __shared__ unsigned s_prefix;
  __shared__ unsigned s_k;
  const int e = blockIdx.x;
  const unsigned* p = (const unsigned*)(probs_et + (size_t)e * TOKENS);
  if (threadIdx.x == 0) { s_prefix = 0u; s_k = CAP; }

  for (int round = 0; round < 4; ++round) {
    const int shift = 24 - 8 * round;
    hist[threadIdx.x] = 0u;
    __syncthreads();
    const unsigned prefix = s_prefix;
    for (int t = threadIdx.x; t < TOKENS; t += 256) {
      unsigned u = p[t];
      if (round == 0 || (u >> (shift + 8)) == prefix)
        atomicAdd(&hist[(u >> shift) & 255u], 1u);
    }
    __syncthreads();
    if (threadIdx.x == 0) {
      unsigned k = s_k, cum = 0u;
      int b = 255;
      for (; b >= 0; --b) { cum += hist[b]; if (cum >= k) break; }
      s_k = k - (cum - hist[b]);          // slots left among ties in bin b
      s_prefix = (prefix << 8) | (unsigned)b;
    }
    __syncthreads();
  }
  if (threadIdx.x == 0) { thr[e] = s_prefix; quota[e] = s_k; }
}

// ---------------------------------------------------------------------------
// Kernel 3: per-expert ordered selection. One block per expert walks tokens in
// order; accepts p > theta plus the first quota ties (lowest index first,
// matching top_k tie-break) via wave32 ballot + block prefix scan.
// ---------------------------------------------------------------------------
__global__ __launch_bounds__(1024) void mask_kernel(
    const float* __restrict__ probs_et, const unsigned* __restrict__ thr,
    const unsigned* __restrict__ quota, float* __restrict__ mask_te)
{
  __shared__ unsigned wave_cnt[32];
  __shared__ unsigned wave_off[32];
  __shared__ unsigned s_base;
  const int e = blockIdx.x;
  const unsigned theta = thr[e];
  const unsigned q     = quota[e];
  const unsigned* p = (const unsigned*)(probs_et + (size_t)e * TOKENS);
  const int tid  = threadIdx.x;
  const int wave = tid >> 5;
  const int lane = tid & 31;
  if (tid == 0) s_base = 0u;
  __syncthreads();

  for (int chunk = 0; chunk < TOKENS; chunk += 1024) {
    const int t = chunk + tid;
    const unsigned u = p[t];
    const bool gt = u > theta;
    const bool eq = (u == theta);
    const unsigned bal = __builtin_amdgcn_ballot_w32(eq);
    const unsigned lanePre = __builtin_popcount(bal & ((1u << lane) - 1u));
    if (lane == 0) wave_cnt[wave] = __builtin_popcount(bal);
    __syncthreads();
    if (tid == 0) {
      unsigned a = s_base;
      for (int w = 0; w < 32; ++w) { wave_off[w] = a; a += wave_cnt[w]; }
      s_base = a;
    }
    __syncthreads();
    const unsigned rank = wave_off[wave] + lanePre;
    const bool take = gt || (eq && rank < q);
    mask_te[(size_t)t * NUM_EXPERTS + e] = take ? 1.0f : 0.0f;
    __syncthreads();
  }
}

// ---------------------------------------------------------------------------
// Kernel 4: per-token combine + normalize; write (weights, mask) outputs.
// ---------------------------------------------------------------------------
__global__ __launch_bounds__(256) void finalize_kernel(
    const float* __restrict__ probs_te, const float* __restrict__ mask_te,
    float* __restrict__ out_w, float* __restrict__ out_m)
{
  const int t = blockIdx.x * 256 + threadIdx.x;
  const float* p = probs_te + (size_t)t * NUM_EXPERTS;
  const float* m = mask_te  + (size_t)t * NUM_EXPERTS;
  float w[NUM_EXPERTS];
  float sum = 0.0f;
#pragma unroll
  for (int e = 0; e < NUM_EXPERTS; e += 4) {
    float4 pv = *(const float4*)(p + e);
    float4 mv = *(const float4*)(m + e);
    w[e]     = pv.x * mv.x;
    w[e + 1] = pv.y * mv.y;
    w[e + 2] = pv.z * mv.z;
    w[e + 3] = pv.w * mv.w;
    sum += w[e] + w[e + 1] + w[e + 2] + w[e + 3];
    *(float4*)(out_m + (size_t)t * NUM_EXPERTS + e) = mv;
  }
  const float inv = 1.0f / (sum + 1e-10f);
#pragma unroll
  for (int e = 0; e < NUM_EXPERTS; e += 4) {
    float4 o;
    o.x = w[e] * inv; o.y = w[e + 1] * inv;
    o.z = w[e + 2] * inv; o.w = w[e + 3] * inv;
    *(float4*)(out_w + (size_t)t * NUM_EXPERTS + e) = o;
  }
}

// ---------------------------------------------------------------------------
extern "C" void kernel_launch(void* const* d_in, const int* in_sizes, int n_in,
                              void* d_out, int out_size, void* d_ws, size_t ws_size,
                              hipStream_t stream) {
  const float* hidden = (const float*)d_in[0];   // [8,4096,2048] f32
  const float* gate   = (const float*)d_in[1];   // [64,2048] f32
  float* out = (float*)d_out;                    // weights then mask, each T*64

  char* ws = (char*)d_ws;
  float*    probs_te = (float*)ws;                                // 8 MiB
  float*    probs_et = (float*)(ws + (size_t)8  * 1024 * 1024);   // 8 MiB
  float*    mask_te  = (float*)(ws + (size_t)16 * 1024 * 1024);   // 8 MiB
  unsigned* thr      = (unsigned*)(ws + (size_t)24 * 1024 * 1024);
  unsigned* quota    = thr + NUM_EXPERTS;

  gemm_softmax_kernel<<<TOKENS / 128, 256, 0, stream>>>(hidden, gate, probs_te, probs_et);
  threshold_kernel<<<NUM_EXPERTS, 256, 0, stream>>>(probs_et, thr, quota);
  mask_kernel<<<NUM_EXPERTS, 1024, 0, stream>>>(probs_et, thr, quota, mask_te);
  finalize_kernel<<<TOKENS / 256, 256, 0, stream>>>(probs_te, mask_te,
                                                    out, out + (size_t)TOKENS * NUM_EXPERTS);
}
